// AttBlock_41257455845892
// MI455X (gfx1250) — matrix operand order repaired and checked by reading, not compiled, not run on hardware
//
#include <hip/hip_runtime.h>
#include <hip/hip_bf16.h>
#include <stdint.h>

// Problem constants (match reference harness)
#define BB   256   // batch
#define LL   256   // seq len
#define DDIM 768   // input size
#define HH   1024  // hidden
#define TT   32    // output steps
#define HOFF 2304  // 3*DDIM : offset of h inside u = [v1 v2 v3 h]
#define KCAT 3328  // 3*DDIM + HH
#define USTR KCAT

typedef __attribute__((ext_vector_type(16))) __bf16 v16bf;
typedef __attribute__((ext_vector_type(8)))  float  v8f;

__device__ __forceinline__ v8f wmma_bf16f32(v16bf a, v16bf b, v8f c) {
  // v_wmma_f32_16x16x32_bf16: D(16x16,f32) = A(16x32,bf16) x B(32x16,bf16) + C
  return __builtin_amdgcn_wmma_f32_16x16x32_bf16(false, a, false, b, (short)0, c,
                                                 false, false);
}

__device__ __forceinline__ unsigned short f2bf(float f) {
  unsigned int u = __float_as_uint(f);
  u += 0x7FFFu + ((u >> 16) & 1u);           // round-to-nearest-even
  return (unsigned short)(u >> 16);
}
__device__ __forceinline__ float bf2f(unsigned short h) {
  return __uint_as_float(((unsigned int)h) << 16);
}
__device__ __forceinline__ float sigf(float x) { return 1.f / (1.f + __expf(-x)); }
__device__ __forceinline__ float tanhfast(float x) {
  x = fminf(fmaxf(x, -15.f), 15.f);
  float e = __expf(2.f * x);
  return (e - 1.f) / (e + 1.f);
}

// A fragment (16x32 bf16, row-major MxK memory, ISA 7.12.2):
// lane m=lane&15, half=lane>>4; elem e -> k = (e&7) + half*8 + (e>>3)*16
// => two contiguous 16B loads at k0+half*8 and k0+half*8+16.
__device__ __forceinline__ v16bf load_a_frag(const unsigned short* p, int stride,
                                             int k0, int lane) {
  const int half = (lane >> 4) & 1;
  const unsigned short* rp = p + (size_t)(lane & 15) * stride + k0 + half * 8;
  union { uint4 q[2]; v16bf v; } u;
  u.q[0] = *(const uint4*)(rp);
  u.q[1] = *(const uint4*)(rp + 16);
  return u.v;
}
// B fragment (32x16 bf16) loaded from row-major NxK memory (i.e. B^T = W rows):
// lane n=lane&15, half=lane>>4; elem e -> k = (e&7) + half*8 + (e>>3)*16
// => two contiguous 16B loads at k0+half*8 and k0+half*8+16.
__device__ __forceinline__ v16bf load_bT_frag(const unsigned short* p, int stride,
                                              int k0, int lane) {
  const int half = (lane >> 4) & 1;
  const unsigned short* rp = p + (size_t)(lane & 15) * stride + k0 + half * 8;
  union { uint4 q[2]; v16bf v; } u;
  u.q[0] = *(const uint4*)(rp);
  u.q[1] = *(const uint4*)(rp + 16);
  return u.v;
}

// ---------------- conversion / packing kernels ----------------
__global__ void k_f32_to_bf16(const float* __restrict__ src,
                              unsigned short* __restrict__ dst, long n) {
  long i = (long)blockIdx.x * blockDim.x + threadIdx.x;
  long st = (long)gridDim.x * blockDim.x;
  for (; i < n; i += st) dst[i] = f2bf(src[i]);
}
// wcat[r][k] = k<HOFF ? W_ih[r][k] : W_hh[r][k-HOFF]   (r in [0,4096), bf16)
__global__ void k_build_wcat(const float* __restrict__ wih,
                             const float* __restrict__ whh,
                             unsigned short* __restrict__ wcat) {
  long n = (long)(4 * HH) * KCAT;
  long i = (long)blockIdx.x * blockDim.x + threadIdx.x;
  long st = (long)gridDim.x * blockDim.x;
  for (; i < n; i += st) {
    long r = i / KCAT, k = i - r * KCAT;
    float v = (k < HOFF) ? wih[r * (long)HOFF + k] : whh[r * (long)HH + (k - HOFF)];
    wcat[i] = f2bf(v);
  }
}
__global__ void k_build_bcat(const float* __restrict__ bih,
                             const float* __restrict__ bhh,
                             float* __restrict__ bcat) {
  int i = blockIdx.x * blockDim.x + threadIdx.x;
  if (i < 4 * HH) bcat[i] = bih[i] + bhh[i];
}

// ---------------- persistent recurrent kernel ----------------
// One workgroup per 16 batch rows; 32 waves; loops all TT steps.
// LDS: u[16][KCAT] bf16 (= [v1 v2 v3 h]), scores[48][256] f32, hnew[16][HH] bf16.
extern "C" __global__ void __launch_bounds__(1024, 1)
k_attlstm(const unsigned short* __restrict__ xb,   // [B][L][D] bf16
          const unsigned short* __restrict__ wh,   // [3][L][H] bf16
          const unsigned short* __restrict__ wcat, // [4H][KCAT] bf16
          const float* __restrict__ h0, const float* __restrict__ c0,
          const float* __restrict__ b1, const float* __restrict__ b2,
          const float* __restrict__ b3, const float* __restrict__ bcat,
          float* __restrict__ out)                 // [B][T][H] f32
{
  extern __shared__ char smem[];
  unsigned short* uL   = (unsigned short*)smem;                      // 16*KCAT
  float*          scL  = (float*)(smem + 16 * KCAT * 2);             // 48*256
  unsigned short* hnL  = (unsigned short*)((char*)scL + 48 * LL * 4);// 16*HH

  const int tid  = threadIdx.x;
  const int lane = tid & 31;
  const int w    = tid >> 5;         // wave 0..31
  const int half = (lane >> 4) & 1;
  const int ln16 = lane & 15;
  const int b0   = blockIdx.x * 16;

  // init h segment of U from h0 (bf16)
  for (int j = tid; j < 16 * HH; j += 1024) {
    int m = j >> 10, k = j & (HH - 1);
    uL[m * USTR + HOFF + k] = f2bf(h0[(size_t)(b0 + m) * HH + k]);
  }
  // c kept in registers: wave w owns h-columns of tiles ht=2w,2w+1
  float creg[2][8];
#pragma unroll
  for (int i = 0; i < 2; ++i) {
    const int col = (2 * w + i) * 16 + ln16;
#pragma unroll
    for (int r = 0; r < 8; ++r)
      creg[i][r] = c0[(size_t)(b0 + r + 8 * half) * HH + col];
  }
  __syncthreads();

  const v8f zero8 = {0.f, 0.f, 0.f, 0.f, 0.f, 0.f, 0.f, 0.f};

  for (int t = 0; t < TT; ++t) {
    // ---- Stage A: scores[head][m][l] = (h @ Wh^T)[m][l] + bias[l] (WMMA) ----
    // Software-pipelined: fragments for k+32 are in flight while WMMA k runs.
    for (int tile = w; tile < 48; tile += 32) {
      const int head = tile >> 4, nt = tile & 15;
      const unsigned short* wp = wh + ((size_t)head * LL + (size_t)nt * 16) * HH;
      v8f acc = zero8;
      v16bf a = load_a_frag(uL, USTR, HOFF, lane);
      v16bf b = load_bT_frag(wp, HH, 0, lane);
      for (int k0 = 0; k0 < HH; k0 += 32) {
        const int kn = (k0 + 32 < HH) ? (k0 + 32) : 0;   // wrap: in-bounds dummy
        v16bf an = load_a_frag(uL, USTR, HOFF + kn, lane);
        v16bf bn = load_bT_frag(wp, HH, kn, lane);
        acc = wmma_bf16f32(a, b, acc);
        a = an; b = bn;
      }
      const float* bias = (head == 0) ? b1 : (head == 1) ? b2 : b3;
      const int l = nt * 16 + ln16;
      const float bl = bias[l];
#pragma unroll
      for (int r = 0; r < 8; ++r)
        scL[(head * 16 + r + 8 * half) * LL + l] = acc[r] + bl;
    }
    __syncthreads();

    // ---- Stage B: row softmax over L=256, in place (wave per row) ----
    for (int row = w; row < 48; row += 32) {
      float* rp = scL + row * LL;
      float v[8], mx = -3.0e38f;
#pragma unroll
      for (int j = 0; j < 8; ++j) { v[j] = rp[lane + 32 * j]; mx = fmaxf(mx, v[j]); }
#pragma unroll
      for (int off = 16; off; off >>= 1) mx = fmaxf(mx, __shfl_xor(mx, off, 32));
      float s = 0.f;
#pragma unroll
      for (int j = 0; j < 8; ++j) { v[j] = __expf(v[j] - mx); s += v[j]; }
#pragma unroll
      for (int off = 16; off; off >>= 1) s += __shfl_xor(s, off, 32);
      const float inv = 1.f / s;
#pragma unroll
      for (int j = 0; j < 8; ++j) rp[lane + 32 * j] = v[j] * inv;
    }
    __syncthreads();

    // ---- Stage C: v[head] = a[head] @ x[b]  (per-batch attend, VALU) ----
    {
      const int bloc = w & 15, dh = w >> 4;    // each wave: one b, half of D
      const int d0 = dh * 384;
      const unsigned short* xp = xb + ((size_t)(b0 + bloc) * LL) * DDIM;
      float acc[3][12];
#pragma unroll
      for (int h3 = 0; h3 < 3; ++h3)
#pragma unroll
        for (int j = 0; j < 12; ++j) acc[h3][j] = 0.f;
      for (int l = 0; l < LL; ++l) {
        const float a0 = scL[(0 * 16 + bloc) * LL + l];
        const float a1 = scL[(1 * 16 + bloc) * LL + l];
        const float a2 = scL[(2 * 16 + bloc) * LL + l];
        const unsigned short* xr = xp + (size_t)l * DDIM + d0 + lane * 4;
#pragma unroll
        for (int j = 0; j < 3; ++j) {
          ushort4 q = *(const ushort4*)(xr + j * 128);
          const float f0 = bf2f(q.x), f1 = bf2f(q.y), f2 = bf2f(q.z), f3 = bf2f(q.w);
          acc[0][j*4+0] = fmaf(a0,f0,acc[0][j*4+0]); acc[0][j*4+1] = fmaf(a0,f1,acc[0][j*4+1]);
          acc[0][j*4+2] = fmaf(a0,f2,acc[0][j*4+2]); acc[0][j*4+3] = fmaf(a0,f3,acc[0][j*4+3]);
          acc[1][j*4+0] = fmaf(a1,f0,acc[1][j*4+0]); acc[1][j*4+1] = fmaf(a1,f1,acc[1][j*4+1]);
          acc[1][j*4+2] = fmaf(a1,f2,acc[1][j*4+2]); acc[1][j*4+3] = fmaf(a1,f3,acc[1][j*4+3]);
          acc[2][j*4+0] = fmaf(a2,f0,acc[2][j*4+0]); acc[2][j*4+1] = fmaf(a2,f1,acc[2][j*4+1]);
          acc[2][j*4+2] = fmaf(a2,f2,acc[2][j*4+2]); acc[2][j*4+3] = fmaf(a2,f3,acc[2][j*4+3]);
        }
      }
#pragma unroll
      for (int h3 = 0; h3 < 3; ++h3)
#pragma unroll
        for (int j = 0; j < 3; ++j) {
          ushort4 q;
          q.x = f2bf(acc[h3][j*4+0]); q.y = f2bf(acc[h3][j*4+1]);
          q.z = f2bf(acc[h3][j*4+2]); q.w = f2bf(acc[h3][j*4+3]);
          *(ushort4*)(uL + (size_t)bloc * USTR + h3 * DDIM + d0 + j * 128 + lane * 4) = q;
        }
    }
    __syncthreads();

    // ---- Stage D: gates = u @ Wcat^T + bcat, then fused LSTM update (WMMA) ----
    // Software-pipelined on A (LDS) and the 4 B fragments (global).
#pragma unroll
    for (int i = 0; i < 2; ++i) {
      const int ht = 2 * w + i;                 // h-tile 0..63
      v8f gacc[4];
#pragma unroll
      for (int g = 0; g < 4; ++g) gacc[g] = zero8;
      const unsigned short* wbase = wcat + (size_t)ht * 16 * KCAT;
      v16bf a = load_a_frag(uL, USTR, 0, lane);
      v16bf bfr[4];
#pragma unroll
      for (int g = 0; g < 4; ++g)
        bfr[g] = load_bT_frag(wbase + (size_t)g * HH * KCAT, KCAT, 0, lane);
      for (int k0 = 0; k0 < KCAT; k0 += 32) {
        const int kn = (k0 + 32 < KCAT) ? (k0 + 32) : 0;  // wrap: in-bounds dummy
        v16bf an = load_a_frag(uL, USTR, kn, lane);
        v16bf bn[4];
#pragma unroll
        for (int g = 0; g < 4; ++g)
          bn[g] = load_bT_frag(wbase + (size_t)g * HH * KCAT, KCAT, kn, lane);
#pragma unroll
        for (int g = 0; g < 4; ++g)
          gacc[g] = wmma_bf16f32(a, bfr[g], gacc[g]);
        a = an;
#pragma unroll
        for (int g = 0; g < 4; ++g) bfr[g] = bn[g];
      }
      const int col = ht * 16 + ln16;
      const float bi = bcat[0 * HH + col], bfg = bcat[1 * HH + col];
      const float bg = bcat[2 * HH + col], bo  = bcat[3 * HH + col];
#pragma unroll
      for (int r = 0; r < 8; ++r) {
        const int m = r + 8 * half;
        const float ig = sigf(gacc[0][r] + bi);
        const float fg = sigf(gacc[1][r] + bfg);
        const float gg = tanhfast(gacc[2][r] + bg);
        const float og = sigf(gacc[3][r] + bo);
        const float cn = fg * creg[i][r] + ig * gg;
        creg[i][r] = cn;
        const float hn = og * tanhfast(cn);
        hnL[m * HH + col] = f2bf(hn);                          // staged (no race on uL)
        out[(((size_t)(b0 + m)) * TT + t) * HH + col] = hn;
      }
    }
    __syncthreads();
    // publish h_new into U's h segment for next step
    {
      const unsigned int* src = (const unsigned int*)hnL;
      unsigned int* dst = (unsigned int*)uL;
      for (int j = tid; j < 16 * HH / 2; j += 1024) {
        const int row = j >> 9;              // /512 (uints per row)
        const int c2  = (j & 511) * 2;       // even bf16 column
        dst[((size_t)row * USTR + HOFF + c2) >> 1] = src[j];
      }
    }
    __syncthreads();
  }
}

// ---------------- host launcher ----------------
extern "C" void kernel_launch(void* const* d_in, const int* in_sizes, int n_in,
                              void* d_out, int out_size, void* d_ws, size_t ws_size,
                              hipStream_t stream) {
  const float* x   = (const float*)d_in[0];
  const float* h0  = (const float*)d_in[1];
  const float* c0  = (const float*)d_in[2];
  const float* W1  = (const float*)d_in[3];
  const float* b1  = (const float*)d_in[4];
  const float* W2  = (const float*)d_in[5];
  const float* b2  = (const float*)d_in[6];
  const float* W3  = (const float*)d_in[7];
  const float* b3  = (const float*)d_in[8];
  const float* Wih = (const float*)d_in[9];
  const float* Whh = (const float*)d_in[10];
  const float* bih = (const float*)d_in[11];
  const float* bhh = (const float*)d_in[12];
  (void)in_sizes; (void)n_in; (void)out_size; (void)ws_size;

  char* ws = (char*)d_ws;
  unsigned short* xb   = (unsigned short*)ws; ws += (size_t)BB * LL * DDIM * 2;
  unsigned short* wh   = (unsigned short*)ws; ws += (size_t)3 * LL * HH * 2;
  unsigned short* wcat = (unsigned short*)ws; ws += (size_t)4 * HH * KCAT * 2;
  float*          bcat = (float*)ws;          ws += (size_t)4 * HH * 4;

  const long nx = (long)BB * LL * DDIM;
  k_f32_to_bf16<<<4096, 256, 0, stream>>>(x, xb, nx);
  const long nw = (long)LL * HH;
  k_f32_to_bf16<<<1024, 256, 0, stream>>>(W1, wh + 0 * nw, nw);
  k_f32_to_bf16<<<1024, 256, 0, stream>>>(W2, wh + 1 * nw, nw);
  k_f32_to_bf16<<<1024, 256, 0, stream>>>(W3, wh + 2 * nw, nw);
  k_build_wcat<<<4096, 256, 0, stream>>>(Wih, Whh, wcat);
  k_build_bcat<<<16, 256, 0, stream>>>(bih, bhh, bcat);

  const size_t lds = (size_t)16 * KCAT * 2 + (size_t)48 * LL * 4 + (size_t)16 * HH * 2;
  hipFuncSetAttribute((const void*)k_attlstm,
                      hipFuncAttributeMaxDynamicSharedMemorySize, (int)lds);
  k_attlstm<<<BB / 16, 1024, lds, stream>>>(xb, wh, wcat, h0, c0, b1, b2, b3, bcat,
                                            (float*)d_out);
}
